// DualVMamba_90271622627933
// MI455X (gfx1250) — compile-verified, matrix operand order checked by reading, not run on hardware
//
#include <hip/hip_runtime.h>
#include <hip/hip_bf16.h>

// ---------------------------------------------------------------------------
// Types for CDNA5 WMMA (wave32, 16x16x32 f16 -> f32 accumulate)
// ---------------------------------------------------------------------------
typedef __attribute__((ext_vector_type(16))) _Float16 v16h;
typedef __attribute__((ext_vector_type(8)))  _Float16 v8h;
typedef __attribute__((ext_vector_type(8)))  float    v8f;

enum { EPI_NONE = 0, EPI_BIAS = 1, EPI_BIAS_SOFTPLUS = 2, EPI_BIAS_ADD = 3 };

// ---------------------------------------------------------------------------
// FAST WMMA GEMM:  C[M,N] = A[M,K] * W[N,K]^T (+ epilogue)
// Preconditions: M % 128 == 0, K % 32 == 0, lda % 4 == 0, ldw % 4 == 0,
// A and W 16-byte aligned. Only N has an edge guard.
// Block: 256 threads = 8 waves (4x2). Each wave computes a 32x32 C tile as
// 2x2 WMMA fragments -> 4 v_wmma per K-step with A/B fragment reuse.
// LDS tiles are fragment-packed (ISA 7.12.2): a row's 32 K-values occupy two
// contiguous 16-half lane slots, so packing is pure b128 load -> b128 store.
// ---------------------------------------------------------------------------
__global__ void __launch_bounds__(256)
gemm_wmma_fast_kernel(const float* __restrict__ A, int lda,
                      const float* __restrict__ W, int ldw,
                      const float* __restrict__ bias,
                      float* __restrict__ C, int ldc,
                      int M, int N, int K, int mode)
{
    __shared__ __align__(32) _Float16 sA[8 * 512];   // 128x32 halves, packed
    __shared__ __align__(32) _Float16 sB[4 * 512];   //  64x32 halves, packed

    const int tid  = threadIdx.x;
    const int lane = tid & 31;
    const int wave = tid >> 5;       // 0..7
    const int wm   = wave >> 1;      // 0..3 -> M sub-block (32 rows)
    const int wn   = wave & 1;       // 0..1 -> N sub-block (32 cols)
    const int row0 = blockIdx.y * 128;
    const int col0 = blockIdx.x * 64;

    // ---- A packing: thread -> (row am, K-half ahf). 16 floats -> 2x v8h.
    const int am    = tid >> 1;              // 0..127
    const int ahf   = tid & 1;               // 0 => k[0..15], 1 => k[16..31]
    const int afrag = am >> 4;
    const int al0   = am & 15;
    _Float16* aDst0 = &sA[afrag * 512 + al0 * 16 + ahf * 8];        // k lo 8
    _Float16* aDst1 = &sA[afrag * 512 + (al0 + 16) * 16 + ahf * 8]; // k hi 8
    const float* aSrc = A + (size_t)(row0 + am) * lda + ahf * 16;

    // ---- B packing: thread -> (row bn, K-quarter bq). 8 floats -> 1x v8h.
    const int bn    = tid >> 2;              // 0..63
    const int bq    = tid & 3;               // k block of 8
    const int bfrag = bn >> 4;
    const int bl    = (bn & 15) + ((bq & 1) << 4);
    const int bj    = (bq >> 1) << 3;
    _Float16* bDst  = &sB[bfrag * 512 + bl * 16 + bj];
    const bool bValid = (col0 + bn) < N;
    const float* bSrc = W + (size_t)(col0 + bn) * ldw + bq * 8;

    v8f acc[2][2];
#pragma unroll
    for (int i = 0; i < 2; ++i)
#pragma unroll
        for (int j = 0; j < 2; ++j)
            acc[i][j] = (v8f){0.f, 0.f, 0.f, 0.f, 0.f, 0.f, 0.f, 0.f};

    for (int k0 = 0; k0 < K; k0 += 32) {
        // stage A (vectorized, branch-free)
        {
            const float4* p = reinterpret_cast<const float4*>(aSrc + k0);
            float4 f0 = p[0], f1 = p[1], f2 = p[2], f3 = p[3];
            v8h h0 = {(_Float16)f0.x, (_Float16)f0.y, (_Float16)f0.z, (_Float16)f0.w,
                      (_Float16)f1.x, (_Float16)f1.y, (_Float16)f1.z, (_Float16)f1.w};
            v8h h1 = {(_Float16)f2.x, (_Float16)f2.y, (_Float16)f2.z, (_Float16)f2.w,
                      (_Float16)f3.x, (_Float16)f3.y, (_Float16)f3.z, (_Float16)f3.w};
            *reinterpret_cast<v8h*>(aDst0) = h0;
            *reinterpret_cast<v8h*>(aDst1) = h1;
        }
        // stage B (N-edge guarded only)
        {
            float4 f0 = {0.f, 0.f, 0.f, 0.f}, f1 = {0.f, 0.f, 0.f, 0.f};
            if (bValid) {
                const float4* p = reinterpret_cast<const float4*>(bSrc + k0);
                f0 = p[0]; f1 = p[1];
            }
            v8h h = {(_Float16)f0.x, (_Float16)f0.y, (_Float16)f0.z, (_Float16)f0.w,
                     (_Float16)f1.x, (_Float16)f1.y, (_Float16)f1.z, (_Float16)f1.w};
            *reinterpret_cast<v8h*>(bDst) = h;
        }
        if (k0 + 32 < K) {
            __builtin_prefetch(aSrc + k0 + 32, 0, 3);
            if (bValid) __builtin_prefetch(bSrc + k0 + 32, 0, 3);
        }
        __syncthreads();

        v16h a0 = *reinterpret_cast<const v16h*>(&sA[(wm * 2 + 0) * 512 + lane * 16]);
        v16h a1 = *reinterpret_cast<const v16h*>(&sA[(wm * 2 + 1) * 512 + lane * 16]);
        v16h b0 = *reinterpret_cast<const v16h*>(&sB[(wn * 2 + 0) * 512 + lane * 16]);
        v16h b1 = *reinterpret_cast<const v16h*>(&sB[(wn * 2 + 1) * 512 + lane * 16]);
        acc[0][0] = __builtin_amdgcn_wmma_f32_16x16x32_f16(false, a0, false, b0, (short)0, acc[0][0], false, false);
        acc[0][1] = __builtin_amdgcn_wmma_f32_16x16x32_f16(false, a0, false, b1, (short)0, acc[0][1], false, false);
        acc[1][0] = __builtin_amdgcn_wmma_f32_16x16x32_f16(false, a1, false, b0, (short)0, acc[1][0], false, false);
        acc[1][1] = __builtin_amdgcn_wmma_f32_16x16x32_f16(false, a1, false, b1, (short)0, acc[1][1], false, false);
        __syncthreads();
    }

    // ---- epilogue: C layout n = lane&15, m = r + 8*(lane>>4)
    const int n     = lane & 15;
    const int mbase = (lane >> 4) * 8;
#pragma unroll
    for (int j = 0; j < 2; ++j) {
        const int gcol = col0 + (wn * 2 + j) * 16 + n;
        if (gcol >= N) continue;
        const float bv = (bias != nullptr) ? bias[gcol] : 0.f;
#pragma unroll
        for (int i = 0; i < 2; ++i) {
            const int rbase = row0 + (wm * 2 + i) * 16 + mbase;
#pragma unroll
            for (int r = 0; r < 8; ++r) {
                float val = acc[i][j][r] + bv;
                if (mode == EPI_BIAS_SOFTPLUS)
                    val = (val > 20.f) ? val : log1pf(__expf(val));
                float* p = &C[(size_t)(rbase + r) * ldc + gcol];
                if (mode == EPI_BIAS_ADD) *p = *p + val;
                else                      *p = val;
            }
        }
    }
}

// ---------------------------------------------------------------------------
// Generic WMMA GEMM (fully guarded) — used only for odd shapes (dt_proj,
// K = dt_rank which is not a multiple of 32).
// ---------------------------------------------------------------------------
__global__ void __launch_bounds__(256)
gemm_wmma_kernel(const float* __restrict__ A, int lda,
                 const float* __restrict__ W, int ldw,
                 const float* __restrict__ bias,
                 float* __restrict__ C, int ldc,
                 int M, int N, int K, int mode)
{
    __shared__ __align__(32) _Float16 sA[4 * 512];
    __shared__ __align__(32) _Float16 sB[2 * 512];

    const int tid  = threadIdx.x;
    const int lane = tid & 31;
    const int wave = tid >> 5;
    const int wm   = wave >> 1;
    const int wn   = wave & 1;
    const int row0 = blockIdx.y * 64;
    const int col0 = blockIdx.x * 32;

    v8f acc = {0.f, 0.f, 0.f, 0.f, 0.f, 0.f, 0.f, 0.f};

    for (int k0 = 0; k0 < K; k0 += 32) {
        for (int i = tid; i < 64 * 32; i += 256) {
            int m = i >> 5, k = i & 31;
            int gm = row0 + m, gk = k0 + k;
            float v = (gm < M && gk < K) ? A[(size_t)gm * lda + gk] : 0.f;
            int frag = m >> 4, mm = m & 15;
            int ln = mm + (((k >> 3) & 1) << 4);
            int j  = ((k >> 4) << 3) | (k & 7);
            sA[frag * 512 + ln * 16 + j] = (_Float16)v;
        }
        for (int i = tid; i < 32 * 32; i += 256) {
            int nn2 = i >> 5, k = i & 31;
            int gn = col0 + nn2, gk = k0 + k;
            float v = (gn < N && gk < K) ? W[(size_t)gn * ldw + gk] : 0.f;
            int frag = nn2 >> 4, nn = nn2 & 15;
            int ln = nn + (((k >> 3) & 1) << 4);
            int j  = ((k >> 4) << 3) | (k & 7);
            sB[frag * 512 + ln * 16 + j] = (_Float16)v;
        }
        __syncthreads();
        v16h a = *reinterpret_cast<const v16h*>(&sA[wm * 512 + lane * 16]);
        v16h b = *reinterpret_cast<const v16h*>(&sB[wn * 512 + lane * 16]);
        acc = __builtin_amdgcn_wmma_f32_16x16x32_f16(false, a, false, b, (short)0, acc, false, false);
        __syncthreads();
    }

    const int n     = lane & 15;
    const int mbase = (lane >> 4) * 8;
    const int gcol  = col0 + wn * 16 + n;
    if (gcol < N) {
        float bv = (bias != nullptr) ? bias[gcol] : 0.f;
        for (int r = 0; r < 8; ++r) {
            int grow = row0 + wm * 16 + mbase + r;
            if (grow < M) {
                float val = acc[r] + bv;
                if (mode == EPI_BIAS_SOFTPLUS)
                    val = (val > 20.f) ? val : log1pf(__expf(val));
                float* p = &C[(size_t)grow * ldc + gcol];
                if (mode == EPI_BIAS_ADD) *p = *p + val;
                else                      *p = val;
            }
        }
    }
}

// ---------------------------------------------------------------------------
// LayerNorm: one block per row, shared-memory tree reduction (wave32-safe).
// ---------------------------------------------------------------------------
__global__ void __launch_bounds__(256)
layernorm_kernel(const float* __restrict__ in, float* __restrict__ out,
                 const float* __restrict__ g, const float* __restrict__ b, int D)
{
    __shared__ float sm[256];
    const int row = blockIdx.x;
    const float* x = in + (size_t)row * D;

    float s = 0.f;
    for (int i = threadIdx.x; i < D; i += 256) s += x[i];
    sm[threadIdx.x] = s; __syncthreads();
    for (int st = 128; st > 0; st >>= 1) {
        if (threadIdx.x < st) sm[threadIdx.x] += sm[threadIdx.x + st];
        __syncthreads();
    }
    float mean = sm[0] / (float)D;
    __syncthreads();

    float vs = 0.f;
    for (int i = threadIdx.x; i < D; i += 256) { float d = x[i] - mean; vs += d * d; }
    sm[threadIdx.x] = vs; __syncthreads();
    for (int st = 128; st > 0; st >>= 1) {
        if (threadIdx.x < st) sm[threadIdx.x] += sm[threadIdx.x + st];
        __syncthreads();
    }
    float rstd = rsqrtf(sm[0] / (float)D + 1e-5f);

    float* o = out + (size_t)row * D;
    for (int i = threadIdx.x; i < D; i += 256)
        o[i] = (x[i] - mean) * rstd * g[i] + b[i];
}

// ---------------------------------------------------------------------------
// Patch embed: 4x4 stride-4 conv, 3->96 channels, channels-last output.
// ---------------------------------------------------------------------------
__global__ void patch_embed_kernel(const float* __restrict__ x,
                                   const float* __restrict__ w,
                                   const float* __restrict__ bias,
                                   float* __restrict__ out,
                                   int Bn, int Hp, int Wp)
{
    int idx = blockIdx.x * blockDim.x + threadIdx.x;
    int total = Bn * Hp * Wp * 96;
    if (idx >= total) return;
    int co = idx % 96;
    int l  = (idx / 96) % (Hp * Wp);
    int bb = idx / (96 * Hp * Wp);
    int ph = l / Wp, pw = l % Wp;
    float acc = bias[co];
    for (int ci = 0; ci < 3; ++ci)
        for (int kh = 0; kh < 4; ++kh)
            for (int kw = 0; kw < 4; ++kw)
                acc += x[((size_t)(bb * 3 + ci) * 256 + (ph * 4 + kh)) * 128 + (pw * 4 + kw)]
                     * w[((co * 3 + ci) * 4 + kh) * 4 + kw];
    out[((size_t)bb * (Hp * Wp) + l) * 96 + co] = acc;
}

// ---------------------------------------------------------------------------
// Causal depthwise conv (d_conv=4) + SiLU. xa = first di channels of xz.
// ---------------------------------------------------------------------------
__global__ void dwconv_silu_kernel(const float* __restrict__ xz,
                                   const float* __restrict__ w,
                                   const float* __restrict__ b,
                                   float* __restrict__ xc,
                                   int Bn, int L, int di)
{
    int idx = blockIdx.x * blockDim.x + threadIdx.x;
    int total = Bn * L * di;
    if (idx >= total) return;
    int c  = idx % di;
    int l  = (idx / di) % L;
    int bb = idx / (di * L);
    const float* xa = xz + (size_t)bb * L * (2 * di);
    float acc = b[c];
    for (int t = 0; t < 4; ++t) {
        int ll = l - 3 + t;
        if (ll >= 0) acc += xa[(size_t)ll * (2 * di) + c] * w[c * 4 + t];
    }
    xc[((size_t)bb * L + l) * di + c] = acc / (1.f + __expf(-acc));
}

// ---------------------------------------------------------------------------
// Selective scan: one thread per (batch, channel); h[16], A[16] in registers.
// ---------------------------------------------------------------------------
__global__ void scan_kernel(const float* __restrict__ dt,
                            const float* __restrict__ dbl,
                            const float* __restrict__ xc,
                            const float* __restrict__ A_log,
                            const float* __restrict__ Dp,
                            float* __restrict__ y,
                            int Bn, int L, int di, int dtr)
{
    int idx = blockIdx.x * blockDim.x + threadIdx.x;
    if (idx >= Bn * di) return;
    int c  = idx % di;
    int bb = idx / di;
    float Areg[16], h[16];
    for (int n = 0; n < 16; ++n) { Areg[n] = -__expf(A_log[c * 16 + n]); h[n] = 0.f; }
    float Dv = Dp[c];
    int sdbl = dtr + 32;
    for (int l = 0; l < L; ++l) {
        size_t r = (size_t)bb * L + l;
        float dtv = dt[r * di + c];
        float xv  = xc[r * di + c];
        const float* Bm = dbl + r * sdbl + dtr;
        const float* Cm = Bm + 16;
        float yv = 0.f;
        for (int n = 0; n < 16; ++n) {
            float dA = __expf(dtv * Areg[n]);
            h[n] = dA * h[n] + dtv * Bm[n] * xv;
            yv  += h[n] * Cm[n];
        }
        y[r * di + c] = yv + xv * Dv;
    }
}

// y *= silu(z)
__global__ void gate_kernel(float* __restrict__ y, const float* __restrict__ xz,
                            int total, int di)
{
    int idx = blockIdx.x * blockDim.x + threadIdx.x;
    if (idx >= total) return;
    int c   = idx % di;
    int row = idx / di;
    float z = xz[(size_t)row * (2 * di) + di + c];
    y[idx] *= z / (1.f + __expf(-z));
}

// Cross-stream fusion with learned gates.
__global__ void fusion_kernel(const float* __restrict__ xr, const float* __restrict__ xi,
                              const float* __restrict__ alpha, const float* __restrict__ beta,
                              float* __restrict__ orr, float* __restrict__ oii,
                              int total, int Dd)
{
    int idx = blockIdx.x * blockDim.x + threadIdx.x;
    if (idx >= total) return;
    int c = idx % Dd;
    float gr = 0.5f * tanhf(alpha[c]) + 0.5f;
    float gi = 0.5f * tanhf(beta[c]) + 0.5f;
    float a = xr[idx], b = xi[idx];
    orr[idx] = (1.f - gr) * a + gr * b;
    oii[idx] = (1.f - gi) * b + gi * a;
}

// Patch-merge rearrange: [B,H,W,C] -> [B,(H/2)(W/2),4C], order 00,10,01,11.
__global__ void merge_rearrange_kernel(const float* __restrict__ x, float* __restrict__ out,
                                       int Bn, int H, int Wd, int C)
{
    int idx = blockIdx.x * blockDim.x + threadIdx.x;
    int H2 = H >> 1, W2 = Wd >> 1;
    int total = Bn * H2 * W2 * 4 * C;
    if (idx >= total) return;
    int ch = idx % (4 * C);
    int l  = (idx / (4 * C)) % (H2 * W2);
    int bb = idx / ((4 * C) * H2 * W2);
    int q = ch / C, c = ch % C;
    int h2 = l / W2, w2 = l % W2;
    int dh = (q == 1 || q == 3) ? 1 : 0;
    int dw = (q >= 2) ? 1 : 0;
    int hh = h2 * 2 + dh, ww = w2 * 2 + dw;
    out[idx] = x[((size_t)bb * H * Wd + hh * Wd + ww) * C + c];
}

// Mean over sequence: in [B,L,D] -> out[b*D + c].
__global__ void mean_rows_kernel(const float* __restrict__ in, float* __restrict__ out,
                                 int Bn, int L, int D)
{
    int idx = blockIdx.x * blockDim.x + threadIdx.x;
    if (idx >= Bn * D) return;
    int c  = idx % D;
    int bb = idx / D;
    float s = 0.f;
    for (int l = 0; l < L; ++l) s += in[((size_t)bb * L + l) * D + c];
    out[(size_t)bb * D + c] = s / (float)L;
}

// Batch-norm head over R=8 feature rows.
__global__ void bn_kernel(const float* __restrict__ feat,
                          const float* __restrict__ g, const float* __restrict__ b,
                          float* __restrict__ out, int R, int D)
{
    int c = blockIdx.x * blockDim.x + threadIdx.x;
    if (c >= D) return;
    float m = 0.f;
    for (int r = 0; r < R; ++r) m += feat[(size_t)r * D + c];
    m /= (float)R;
    float v = 0.f;
    for (int r = 0; r < R; ++r) { float d = feat[(size_t)r * D + c] - m; v += d * d; }
    v /= (float)R;
    float rs = rsqrtf(v + 1e-5f);
    for (int r = 0; r < R; ++r)
        out[(size_t)r * D + c] = (feat[(size_t)r * D + c] - m) * rs * g[c] + b[c];
}

// ---------------------------------------------------------------------------
// Host-side parameter bookkeeping (jax tree_flatten order: dict keys sorted).
// ---------------------------------------------------------------------------
struct BlockP {
    const float *A_log, *D, *conv_b, *conv_w, *dt_b, *dt_w, *in_b, *in_w,
                *ln_b, *ln_g, *out_b, *out_w, *xproj_w;
};
struct StageP {
    BlockP blocks[9];
    const float *ds_norm_b, *ds_norm_g, *ds_w;
};
struct StreamP {
    const float *norm_b, *norm_g, *pe_b, *pe_ln_b, *pe_ln_g, *pe_w;
    StageP st[4];
};

static void parse_stream(void* const* d_in, int& idx, StreamP& s)
{
    auto nxt = [&]() { return (const float*)d_in[idx++]; };
    s.norm_b = nxt(); s.norm_g = nxt();
    s.pe_b = nxt(); s.pe_ln_b = nxt(); s.pe_ln_g = nxt(); s.pe_w = nxt();
    static const int deps[4] = {2, 2, 9, 2};
    for (int t = 0; t < 4; ++t) {
        for (int j = 0; j < deps[t]; ++j) {
            BlockP& bp = s.st[t].blocks[j];
            bp.A_log = nxt(); bp.D = nxt(); bp.conv_b = nxt(); bp.conv_w = nxt();
            bp.dt_b = nxt();  bp.dt_w = nxt(); bp.in_b = nxt(); bp.in_w = nxt();
            bp.ln_b = nxt();  bp.ln_g = nxt(); bp.out_b = nxt(); bp.out_w = nxt();
            bp.xproj_w = nxt();
        }
        if (t < 3) { s.st[t].ds_norm_b = nxt(); s.st[t].ds_norm_g = nxt(); s.st[t].ds_w = nxt(); }
        else { s.st[t].ds_norm_b = s.st[t].ds_norm_g = s.st[t].ds_w = nullptr; }
    }
}

extern "C" void kernel_launch(void* const* d_in, const int* in_sizes, int n_in,
                              void* d_out, int out_size, void* d_ws, size_t ws_size,
                              hipStream_t stream)
{
    (void)in_sizes; (void)n_in; (void)out_size; (void)ws_size;

    const float* x1   = (const float*)d_in[0];
    const float* x2   = (const float*)d_in[1];
    const float* bn_b = (const float*)d_in[2];
    const float* bn_g = (const float*)d_in[3];
    const float* fus_alpha[4];
    const float* fus_beta[4];
    int idx = 4;
    for (int s = 0; s < 4; ++s) { fus_alpha[s] = (const float*)d_in[idx++]; fus_beta[s] = (const float*)d_in[idx++]; }
    StreamP ir, rgb;
    parse_stream(d_in, idx, ir);   // 'ir' sorts before 'rgb'
    parse_stream(d_in, idx, rgb);

    // Workspace carve (bump allocator, float units).
    float* ws = (float*)d_ws;
    size_t off = 0;
    auto alloc = [&](size_t n) { float* p = ws + off; off += n; return p; };
    float* bufAr = alloc(786432);
    float* bufAi = alloc(786432);
    float* bufBr = alloc(786432);
    float* bufBi = alloc(786432);
    float* lnb   = alloc(786432);
    float* xz    = alloc(3145728);
    float* xc    = alloc(1572864);
    float* dbl   = alloc(524288);
    float* dtb   = alloc(1572864);
    float* yb    = alloc(1572864);
    float* mg    = alloc(786432);
    float* lnf   = alloc(131072);

    const int kDims[4]   = {96, 192, 384, 768};
    const int kDepths[4] = {2, 2, 9, 2};
    const int Bn = 4;

    auto gemm = [&](const float* A, int lda, const float* W, int ldw, const float* bias,
                    float* C, int ldc, int M, int N, int K, int mode) {
        bool fast = (M % 128 == 0) && (K % 32 == 0) && (lda % 4 == 0) && (ldw % 4 == 0);
        if (fast) {
            dim3 grid((N + 63) / 64, M / 128);
            gemm_wmma_fast_kernel<<<grid, 256, 0, stream>>>(A, lda, W, ldw, bias, C, ldc, M, N, K, mode);
        } else {
            dim3 grid((N + 31) / 32, (M + 63) / 64);
            gemm_wmma_kernel<<<grid, 256, 0, stream>>>(A, lda, W, ldw, bias, C, ldc, M, N, K, mode);
        }
    };
    auto lnorm = [&](const float* in, float* out, const float* g, const float* b,
                     int rows, int D) {
        layernorm_kernel<<<rows, 256, 0, stream>>>(in, out, g, b, D);
    };

    // ---- patch embed ----
    {
        int Hp = 64, Wp = 32, L = Hp * Wp;
        int tot = Bn * L * 96;
        patch_embed_kernel<<<(tot + 255) / 256, 256, 0, stream>>>(x1, rgb.pe_w, rgb.pe_b, lnb, Bn, Hp, Wp);
        lnorm(lnb, bufAr, rgb.pe_ln_g, rgb.pe_ln_b, Bn * L, 96);
        patch_embed_kernel<<<(tot + 255) / 256, 256, 0, stream>>>(x2, ir.pe_w, ir.pe_b, lnb, Bn, Hp, Wp);
        lnorm(lnb, bufAi, ir.pe_ln_g, ir.pe_ln_b, Bn * L, 96);
    }

    float* cur_r = bufAr; float* cur_i = bufAi;
    float* alt_r = bufBr; float* alt_i = bufBi;
    int H = 64, Wd = 32;

    for (int s = 0; s < 4; ++s) {
        const int d   = kDims[s];
        const int L   = H * Wd;
        const int di  = 2 * d;
        const int dtr = (d + 15) / 16;
        const int M   = Bn * L;

        StreamP* sps[2] = {&rgb, &ir};
        float*   xs[2]  = {cur_r, cur_i};
        for (int t = 0; t < 2; ++t) {
            StreamP& sp = *sps[t];
            float* x = xs[t];
            for (int blk = 0; blk < kDepths[s]; ++blk) {
                const BlockP& bp = sp.st[s].blocks[blk];
                lnorm(x, lnb, bp.ln_g, bp.ln_b, M, d);
                // in_proj: [M,d] x [2di,d]^T
                gemm(lnb, d, bp.in_w, d, bp.in_b, xz, 2 * di, M, 2 * di, d, EPI_BIAS);
                // causal depthwise conv + silu
                {
                    int tot = M * di;
                    dwconv_silu_kernel<<<(tot + 255) / 256, 256, 0, stream>>>(
                        xz, bp.conv_w, bp.conv_b, xc, Bn, L, di);
                }
                // x_proj: [M,di] x [dtr+32,di]^T
                gemm(xc, di, bp.xproj_w, di, nullptr, dbl, dtr + 32, M, dtr + 32, di, EPI_NONE);
                // dt_proj + softplus: [M,dtr] x [di,dtr]^T  (generic path, lda=dtr+32)
                gemm(dbl, dtr + 32, bp.dt_w, dtr, bp.dt_b, dtb, di, M, di, dtr, EPI_BIAS_SOFTPLUS);
                // selective scan
                {
                    int tot = Bn * di;
                    scan_kernel<<<(tot + 255) / 256, 256, 0, stream>>>(
                        dtb, dbl, xc, bp.A_log, bp.D, yb, Bn, L, di, dtr);
                }
                // gate with silu(z)
                {
                    int tot = M * di;
                    gate_kernel<<<(tot + 255) / 256, 256, 0, stream>>>(yb, xz, tot, di);
                }
                // out_proj with residual add: x += y @ out_w^T + out_b
                gemm(yb, di, bp.out_w, di, bp.out_b, x, d, M, d, di, EPI_BIAS_ADD);
            }
        }

        // cross-stream fusion
        {
            int tot = M * d;
            fusion_kernel<<<(tot + 255) / 256, 256, 0, stream>>>(
                cur_r, cur_i, fus_alpha[s], fus_beta[s], alt_r, alt_i, tot, d);
            float* t0 = cur_r; cur_r = alt_r; alt_r = t0;
            float* t1 = cur_i; cur_i = alt_i; alt_i = t1;
        }

        // patch merging downsample
        if (s < 3) {
            int L4 = L / 4;
            StreamP* sp2[2] = {&rgb, &ir};
            float* src[2] = {cur_r, cur_i};
            float* dst[2] = {alt_r, alt_i};
            for (int t = 0; t < 2; ++t) {
                int tot = Bn * L4 * 4 * d;
                merge_rearrange_kernel<<<(tot + 255) / 256, 256, 0, stream>>>(
                    src[t], mg, Bn, H, Wd, d);
                lnorm(mg, lnb, sp2[t]->st[s].ds_norm_g, sp2[t]->st[s].ds_norm_b, Bn * L4, 4 * d);
                gemm(lnb, 4 * d, sp2[t]->st[s].ds_w, 4 * d, nullptr, dst[t], 2 * d,
                     Bn * L4, 2 * d, 4 * d, EPI_NONE);
            }
            float* t0 = cur_r; cur_r = alt_r; alt_r = t0;
            float* t1 = cur_i; cur_i = alt_i; alt_i = t1;
            H >>= 1; Wd >>= 1;
        }
    }

    // ---- head: final LN -> mean over L -> feat, then batch-norm ----
    float* outF = (float*)d_out;          // feat: 8x768, then bn: 8x768
    const int Lf = H * Wd;                // 32
    lnorm(cur_r, lnf, rgb.norm_g, rgb.norm_b, Bn * Lf, 768);
    {
        int tot = Bn * 768;
        mean_rows_kernel<<<(tot + 255) / 256, 256, 0, stream>>>(lnf, outF, Bn, Lf, 768);
    }
    lnorm(cur_i, lnf, ir.norm_g, ir.norm_b, Bn * Lf, 768);
    {
        int tot = Bn * 768;
        mean_rows_kernel<<<(tot + 255) / 256, 256, 0, stream>>>(lnf, outF + (size_t)Bn * 768, Bn, Lf, 768);
    }
    bn_kernel<<<(768 + 255) / 256, 256, 0, stream>>>(outF, bn_g, bn_b,
                                                     outF + (size_t)8 * 768, 8, 768);
}
